// SelfAttention_38259568673589
// MI455X (gfx1250) — matrix operand order repaired and checked by reading, not compiled, not run on hardware
//
#include <hip/hip_runtime.h>

typedef __attribute__((ext_vector_type(16))) __bf16 bf16x16;
typedef __attribute__((ext_vector_type(8)))  __bf16 bf16x8;
typedef __attribute__((ext_vector_type(8)))  float  f32x8;
typedef int v4i __attribute__((vector_size(16)));

#if defined(__has_builtin)
#if __has_builtin(__builtin_amdgcn_global_load_async_to_lds_b128) && \
    __has_builtin(__builtin_amdgcn_s_wait_asynccnt)
#define USE_ASYNC_LDS 1
#endif
#endif

#define NB 2
#define NC 256
#define NSP 32768     /* n = D*H*W */
#define NH 4
#define NGROUPS 4
#define GSIZE 64
#define EPSV 1e-5f

/* workspace layout (byte offsets) */
#define OFF_STATS   0u          /* 16 floats: (sum,sumsq) per (b,g)      */
#define OFF_CTX     4096u       /* B*H*64*64 f32 = 131072 B (atomic acc) */
#define OFF_ACOEF   135168u     /* B*C floats                            */
#define OFF_BCOEF   137216u     /* B*C floats                            */
#define OFF_CTXT    139264u     /* B*H*64*64 bf16 (transposed context)   */
#define OFF_WQKV    204800u     /* 768*256 bf16                          */
#define OFF_WOUT    598016u     /* 256*256 bf16                          */
#define OFF_NORM    1048576u    /* B*C*n bf16 = 33.5 MB                  */
#define OFF_QKV     34603008u   /* B*768*n bf16 = 100.7 MB               */
#define OFF_ATTN    135266304u  /* B*C*n bf16 = 33.5 MB                  */
#define ZERO_FLOATS 33792       /* zero first 135168 bytes each launch   */

__device__ __forceinline__ __bf16 bff(float f) {
  unsigned u = __builtin_bit_cast(unsigned, f);
  unsigned r = (u + 0x7FFFu + ((u >> 16) & 1u)) >> 16;
  return __builtin_bit_cast(__bf16, (unsigned short)r);
}
__device__ __forceinline__ float bf2f(__bf16 h) {
  unsigned short s = __builtin_bit_cast(unsigned short, h);
  return __builtin_bit_cast(float, (unsigned)s << 16);
}

__global__ void k_zero(float* p, int n) {
  int i = blockIdx.x * 256 + threadIdx.x;
  if (i < n) p[i] = 0.0f;
}

/* one block per (b,c) channel; atomic partial sums per (b,group) */
__global__ void k_gn_partial(const float* __restrict__ x, float* __restrict__ stats) {
  __shared__ float s1[256], s2[256];
  int bc = blockIdx.x;
  int b = bc / NC, c = bc % NC, g = c / GSIZE;
  const float* row = x + (size_t)bc * NSP;
  float sum = 0.f, sq = 0.f;
  for (int i = threadIdx.x; i < NSP; i += 256) { float v = row[i]; sum += v; sq += v * v; }
  s1[threadIdx.x] = sum; s2[threadIdx.x] = sq; __syncthreads();
  for (int o = 128; o > 0; o >>= 1) {
    if ((int)threadIdx.x < o) { s1[threadIdx.x] += s1[threadIdx.x + o]; s2[threadIdx.x] += s2[threadIdx.x + o]; }
    __syncthreads();
  }
  if (threadIdx.x == 0) {
    atomicAdd(&stats[(b * NGROUPS + g) * 2 + 0], s1[0]);
    atomicAdd(&stats[(b * NGROUPS + g) * 2 + 1], s2[0]);
  }
}

__global__ void k_gn_coef(const float* __restrict__ stats, const float* __restrict__ gamma,
                          const float* __restrict__ beta, float* __restrict__ ac, float* __restrict__ bc_) {
  int b = blockIdx.x, c = threadIdx.x, g = c / GSIZE;
  float cnt = (float)GSIZE * (float)NSP;
  float mean = stats[(b * NGROUPS + g) * 2 + 0] / cnt;
  float var  = stats[(b * NGROUPS + g) * 2 + 1] / cnt - mean * mean;
  float inv = rsqrtf(var + EPSV);
  float a = gamma[c] * inv;
  ac[b * NC + c] = a;
  bc_[b * NC + c] = beta[c] - mean * a;
}

__global__ void k_cast_w(const float* __restrict__ wqkv, const float* __restrict__ wout,
                         __bf16* __restrict__ wq16, __bf16* __restrict__ wo16) {
  int i = blockIdx.x * 256 + threadIdx.x;
  if (i < 3 * NC * NC) wq16[i] = bff(wqkv[i]);
  if (i < NC * NC)     wo16[i] = bff(wout[i]);
}

__global__ void k_norm(const float* __restrict__ x, const float* __restrict__ ac,
                       const float* __restrict__ bc_, __bf16* __restrict__ y) {
  size_t i = (size_t)blockIdx.x * 256 + threadIdx.x;
  if (i >= (size_t)NB * NC * NSP) return;
  int bci = (int)(i / NSP);
  y[i] = bff(x[i] * ac[bci] + bc_[bci]);
}

/* Generic bf16 WMMA GEMM: out[z][outRow+m][n] = sum_k A_z[m][k] * X[xRow+k][n]
   Block tile: 64(M) x 128(N), K step 32.  8 waves, wave w owns N-tile w, all 4 M-tiles.
   A tile: LDS row-major 64x32.  X tile: LDS row-major 32x128 (async global->LDS
   when available), B fragments read with 16 strided ds_load_u16 per lane. */
template<bool OUT_F32, bool HAS_BIAS, bool HAS_RES>
__global__ __launch_bounds__(256)
void k_gemm(const __bf16* __restrict__ A, const __bf16* __restrict__ X,
            float* __restrict__ outF, __bf16* __restrict__ outB,
            const float* __restrict__ bias, const float* __restrict__ res,
            int M, int K, int aZStride, int zDiv, int xOuter, int xInner,
            int outOuter, int outInner)
{
  __shared__ __bf16 Alds[64 * 32];
  __shared__ __bf16 Xl[32 * 128];   /* row-major: Xl[k][col] */
  const int t = threadIdx.x;
  const int lane = t & 31, w = t >> 5;
  const int n0 = blockIdx.x * 128;
  const int mBase = blockIdx.y * 64;
  const int z = blockIdx.z;
  const int xRowBase   = (z / zDiv) * xOuter   + (z % zDiv) * xInner;
  const int outRowBase = (z / zDiv) * outOuter + (z % zDiv) * outInner;
  const __bf16* Az = A + (size_t)z * aZStride;

  f32x8 acc[4];
#pragma unroll
  for (int m = 0; m < 4; m++)
#pragma unroll
    for (int r = 0; r < 8; r++) acc[m][r] = 0.0f;

  const int lrow  = lane & 15;
  const int koffA = (lane < 16) ? 0 : 8;    /* A frag K base (pairs 0..7 then +16) */
  const int kbB   = (lane < 16) ? 0 : 8;    /* B frag K base (pairs 0..7 then +16) */
  const int bcol  = w * 16 + lrow;

  const int arow = t >> 2, acg = (t & 3) * 8;   /* A tile: 16B per thread  */
  const int xkk  = t >> 3, xcg = (t & 7) * 16;  /* X tile: 32B per thread  */

  for (int ks = 0; ks < K; ks += 32) {
    const __bf16* asrc = &Az[(size_t)(mBase + arow) * K + ks + acg];
    const __bf16* xsrc = &X[(size_t)(xRowBase + ks + xkk) * NSP + n0 + xcg];
    __bf16* adst = &Alds[arow * 32 + acg];
    __bf16* xdst = &Xl[xkk * 128 + xcg];
#if defined(USE_ASYNC_LDS)
    __builtin_amdgcn_global_load_async_to_lds_b128((v4i*)asrc, (v4i*)adst, 0, 0);
    __builtin_amdgcn_global_load_async_to_lds_b128((v4i*)xsrc, (v4i*)xdst, 0, 0);
    __builtin_amdgcn_global_load_async_to_lds_b128((v4i*)xsrc, (v4i*)xdst, 16, 0);
    __builtin_amdgcn_s_wait_asynccnt(0);
#else
    *(bf16x8*)adst = *(const bf16x8*)asrc;
    bf16x8 v0 = *(const bf16x8*)xsrc;
    bf16x8 v1 = *(const bf16x8*)(xsrc + 8);
    *(bf16x8*)xdst = v0;
    *(bf16x8*)(xdst + 8) = v1;
#endif
    __syncthreads();

    bf16x16 bfr;
#pragma unroll
    for (int j = 0; j < 16; j++) bfr[j] = Xl[(kbB + j + (j >= 8 ? 8 : 0)) * 128 + bcol];

#pragma unroll
    for (int mt = 0; mt < 4; mt++) {
      bf16x16 afr;
      const __bf16* p = &Alds[(mt * 16 + lrow) * 32 + koffA];
      bf16x8 lo = *(const bf16x8*)p;
      bf16x8 hi = *(const bf16x8*)(p + 16);
#pragma unroll
      for (int i = 0; i < 8; i++) { afr[i] = lo[i]; afr[8 + i] = hi[i]; }
      acc[mt] = __builtin_amdgcn_wmma_f32_16x16x32_bf16(
          false, afr, false, bfr, (short)0, acc[mt], false, false);
    }
    __syncthreads();
  }

  const int hiHalf = (lane >> 4) & 1;
  const int ncol = n0 + w * 16 + lrow;
#pragma unroll
  for (int mt = 0; mt < 4; mt++) {
#pragma unroll
    for (int r = 0; r < 8; r++) {
      int mg = mBase + mt * 16 + r + 8 * hiHalf;
      size_t oi = (size_t)(outRowBase + mg) * NSP + ncol;
      float v = acc[mt][r];
      if (HAS_BIAS) v += bias[mg];
      if (HAS_RES)  v += res[oi];
      if (OUT_F32) outF[oi] = v; else outB[oi] = bff(v);
    }
  }
}

/* softmax over n for each K row (in place, bf16) */
__global__ void k_softmax(__bf16* __restrict__ qkv) {
  __shared__ float red[256];
  int b = blockIdx.x / NC, c = blockIdx.x % NC;
  __bf16* row = qkv + (size_t)(b * 3 * NC + NC + c) * NSP;
  float m = -1e30f;
  for (int i = threadIdx.x; i < NSP; i += 256) m = fmaxf(m, bf2f(row[i]));
  red[threadIdx.x] = m; __syncthreads();
  for (int o = 128; o > 0; o >>= 1) {
    if ((int)threadIdx.x < o) red[threadIdx.x] = fmaxf(red[threadIdx.x], red[threadIdx.x + o]);
    __syncthreads();
  }
  m = red[0]; __syncthreads();
  float s = 0.f;
  for (int i = threadIdx.x; i < NSP; i += 256) s += __expf(bf2f(row[i]) - m);
  red[threadIdx.x] = s; __syncthreads();
  for (int o = 128; o > 0; o >>= 1) {
    if ((int)threadIdx.x < o) red[threadIdx.x] += red[threadIdx.x + o];
    __syncthreads();
  }
  float inv = 1.0f / red[0];
  for (int i = threadIdx.x; i < NSP; i += 256) row[i] = bff(__expf(bf2f(row[i]) - m) * inv);
}

/* context[d][e] = sum_n k[d][n] v[e][n]; K-dim = n, both operands contiguous per lane
   -> direct global fragment loads, no LDS.  grid (B*H, KSPLIT), 16 waves/block */
__global__ __launch_bounds__(512)
void k_context(const __bf16* __restrict__ qkv, float* __restrict__ ctx) {
  int bh = blockIdx.x;
  int b = bh >> 2, h = bh & 3;
  int t = threadIdx.x, lane = t & 31, w = t >> 5;
  int dt = w >> 2, et = w & 3;
  int lrow = lane & 15;
  int koffA = (lane < 16) ? 0 : 8;
  int kbB   = (lane < 16) ? 0 : 8;
  const __bf16* kr = qkv + (size_t)(b * 768 + 256 + h * 64 + dt * 16 + lrow) * NSP;
  const __bf16* vr = qkv + (size_t)(b * 768 + 512 + h * 64 + et * 16 + lrow) * NSP;
  int nBeg = blockIdx.y * (NSP / 8);
  f32x8 acc;
#pragma unroll
  for (int r = 0; r < 8; r++) acc[r] = 0.0f;
  for (int nk = nBeg; nk < nBeg + NSP / 8; nk += 32) {
    /* stream-ahead prefetch (global_prefetch_b8) for the two HBM streams */
    __builtin_prefetch(kr + nk + 2048, 0, 0);
    __builtin_prefetch(vr + nk + 2048, 0, 0);
    bf16x16 afr, bfr;
    bf16x8 alo = *(const bf16x8*)&kr[nk + koffA];
    bf16x8 ahi = *(const bf16x8*)&kr[nk + koffA + 16];
    bf16x8 blo = *(const bf16x8*)&vr[nk + kbB];
    bf16x8 bhi = *(const bf16x8*)&vr[nk + kbB + 16];
#pragma unroll
    for (int i = 0; i < 8; i++) { afr[i] = alo[i]; afr[8 + i] = ahi[i]; bfr[i] = blo[i]; bfr[8 + i] = bhi[i]; }
    acc = __builtin_amdgcn_wmma_f32_16x16x32_bf16(
        false, afr, false, bfr, (short)0, acc, false, false);
  }
  int hiHalf = (lane >> 4) & 1;
#pragma unroll
  for (int r = 0; r < 8; r++) {
    int d = dt * 16 + r + 8 * hiHalf, e = et * 16 + lrow;
    atomicAdd(&ctx[((size_t)bh * 64 + d) * 64 + e], acc[r]);
  }
}

__global__ void k_ctxT(const float* __restrict__ ctx, __bf16* __restrict__ ctT) {
  int i = blockIdx.x * 256 + threadIdx.x;   /* < 2*4*64*64 = 32768 */
  int e = i & 63, d = (i >> 6) & 63, bh = i >> 12;
  ctT[((size_t)bh * 64 + e) * 64 + d] = bff(ctx[((size_t)bh * 64 + d) * 64 + e]);
}

extern "C" void kernel_launch(void* const* d_in, const int* in_sizes, int n_in,
                              void* d_out, int out_size, void* d_ws, size_t ws_size,
                              hipStream_t stream) {
  (void)in_sizes; (void)n_in; (void)out_size; (void)ws_size;
  const float* x     = (const float*)d_in[0];
  const float* gamma = (const float*)d_in[1];
  const float* beta  = (const float*)d_in[2];
  const float* wqkv  = (const float*)d_in[3];
  const float* wout  = (const float*)d_in[4];
  const float* bout  = (const float*)d_in[5];
  float* out = (float*)d_out;
  char* ws = (char*)d_ws;
  float*  stats  = (float*)(ws + OFF_STATS);
  float*  ctx    = (float*)(ws + OFF_CTX);
  float*  ac     = (float*)(ws + OFF_ACOEF);
  float*  bc_    = (float*)(ws + OFF_BCOEF);
  __bf16* ctT    = (__bf16*)(ws + OFF_CTXT);
  __bf16* wq16   = (__bf16*)(ws + OFF_WQKV);
  __bf16* wo16   = (__bf16*)(ws + OFF_WOUT);
  __bf16* norm16 = (__bf16*)(ws + OFF_NORM);
  __bf16* qkv16  = (__bf16*)(ws + OFF_QKV);
  __bf16* attn16 = (__bf16*)(ws + OFF_ATTN);

  k_zero<<<(ZERO_FLOATS + 255) / 256, 256, 0, stream>>>((float*)ws, ZERO_FLOATS);
  k_gn_partial<<<NB * NC, 256, 0, stream>>>(x, stats);
  k_gn_coef<<<NB, NC, 0, stream>>>(stats, gamma, beta, ac, bc_);
  k_cast_w<<<(3 * NC * NC + 255) / 256, 256, 0, stream>>>(wqkv, wout, wq16, wo16);
  k_norm<<<(NB * NC * NSP + 255) / 256, 256, 0, stream>>>(x, ac, bc_, norm16);

  /* qkv = Wqkv(768x256) @ normed(b,256,n)  -> bf16 */
  dim3 g1(NSP / 128, 768 / 64, NB);
  k_gemm<false, false, false><<<g1, 256, 0, stream>>>(
      wq16, norm16, nullptr, qkv16, nullptr, nullptr,
      768, 256, 0, 1, 256, 0, 768, 0);

  k_softmax<<<NB * NC, 256, 0, stream>>>(qkv16);
  k_context<<<dim3(NB * NH, 8), 512, 0, stream>>>(qkv16, ctx);
  k_ctxT<<<128, 256, 0, stream>>>(ctx, ctT);

  /* attn_out = ctxT(64x64) @ q(64,n) per (b,h) -> bf16 */
  dim3 g2(NSP / 128, 1, NB * NH);
  k_gemm<false, false, false><<<g2, 256, 0, stream>>>(
      ctT, qkv16, nullptr, attn16, nullptr, nullptr,
      64, 64, 64 * 64, 4, 768, 64, 256, 64);

  /* final = Wout(256x256) @ attn_out(b,256,n) + bias + residual -> f32 d_out */
  dim3 g3(NSP / 128, 256 / 64, NB);
  k_gemm<true, true, true><<<g3, 256, 0, stream>>>(
      wo16, attn16, out, nullptr, bout, x,
      256, 256, 0, 1, 256, 0, 256, 0);
}
